// EPMoE_37280316129648
// MI455X (gfx1250) — matrix operand order repaired and checked by reading, not compiled, not run on hardware
//
#include <hip/hip_runtime.h>

typedef __attribute__((ext_vector_type(16))) __bf16 bf16x16;
typedef __attribute__((ext_vector_type(8)))  __bf16 bf16x8;
typedef __attribute__((ext_vector_type(4)))  __bf16 bf16x4;
typedef __attribute__((ext_vector_type(8)))  float  v8f;

#define N_TOKENS 2048
#define HIDDEN   1024
#define INTERSZ  2048
#define NEXP     8
#define NROWS    4096          // TOKENS * TOP_K
#define MAXTILES 40            // sum_e ceil(cnt_e/128) <= 39

union FragU { bf16x16 v; bf16x8 h[2]; uint4 u[2]; };

// ---------------------------------------------------------------- CDNA5 async global->LDS helpers
// Copies 32B per lane (two b128 async transfers). data goes from global saddr+vaddr to the LDS address in vdst,
// tracked by ASYNCcnt. offset: applies to both LDS and global addresses.
__device__ __forceinline__ void async_copy32(unsigned lds, unsigned goff, const void* base) {
    asm volatile("global_load_async_to_lds_b128 %0, %1, %2\n\t"
                 "global_load_async_to_lds_b128 %0, %1, %2 offset:16"
                 :: "v"(lds), "v"(goff), "s"(base) : "memory");
}
__device__ __forceinline__ void wait_async0() {
    asm volatile("s_wait_asynccnt 0x0" ::: "memory");
}
__device__ __forceinline__ unsigned lds_addr(const void* p) {
    return (unsigned)(unsigned long long)p;   // LDS aperture: addr[31:0] = LDS byte offset
}

// ---------------------------------------------------------------- transpose + convert f32 -> bf16
// in: [E][R][C] f32  ->  out: [E][C][R] bf16
__global__ __launch_bounds__(256) void transpose_cvt_kernel(const float* __restrict__ in,
                                                            __bf16* __restrict__ out,
                                                            int R, int C) {
    __shared__ __bf16 tile[32][33];
    int e = blockIdx.z;
    int c0 = blockIdx.x * 32, r0 = blockIdx.y * 32;
    const float* src = in + (size_t)e * R * C;
    __bf16* dst = out + (size_t)e * C * R;
    int tx = threadIdx.x & 31;
    int ty = (threadIdx.x >> 5) * 4;
    #pragma unroll
    for (int j = 0; j < 4; ++j)
        tile[ty + j][tx] = (__bf16)src[(size_t)(r0 + ty + j) * C + c0 + tx];
    __syncthreads();
    #pragma unroll
    for (int j = 0; j < 4; ++j)
        dst[(size_t)(c0 + ty + j) * R + r0 + tx] = tile[tx][ty + j];
}

// ---------------------------------------------------------------- routing (1 block, 256 threads)
__global__ __launch_bounds__(256) void routing_kernel(
    const float* __restrict__ logits,      // [T][8]
    float* __restrict__ topk_w,            // [T*2]
    int* __restrict__ src_row,             // [4096] sorted -> flat pair
    int* __restrict__ dest_row,            // [4096] flat pair -> sorted
    int* __restrict__ tile_expert, int* __restrict__ tile_rs,
    int* __restrict__ tile_re, int* __restrict__ num_tiles) {
    __shared__ int s_exp[NROWS];
    __shared__ int s_hist[256][NEXP];
    __shared__ int s_cnt[NEXP];
    __shared__ int s_off[NEXP + 1];
    int tid = threadIdx.x;

    for (int t = tid; t < N_TOKENS; t += 256) {
        float b0 = -1e30f, b1 = -1e30f; int i0 = 0, i1 = 0;
        #pragma unroll
        for (int j = 0; j < NEXP; ++j) {
            float l = logits[t * NEXP + j];
            if (l > b0) { b1 = b0; i1 = i0; b0 = l; i0 = j; }
            else if (l > b1) { b1 = l; i1 = j; }
        }
        float e1 = __expf(b1 - b0);
        float inv = 1.0f / (1.0f + e1);
        topk_w[t * 2 + 0] = inv;
        topk_w[t * 2 + 1] = e1 * inv;
        s_exp[t * 2 + 0] = i0;
        s_exp[t * 2 + 1] = i1;
    }
    __syncthreads();

    int h[NEXP];
    #pragma unroll
    for (int e = 0; e < NEXP; ++e) h[e] = 0;
    int base = tid * 16;
    for (int j = 0; j < 16; ++j) h[s_exp[base + j]]++;
    #pragma unroll
    for (int e = 0; e < NEXP; ++e) s_hist[tid][e] = h[e];
    __syncthreads();

    if (tid < NEXP) {
        int run = 0;
        for (int c = 0; c < 256; ++c) { int v = s_hist[c][tid]; s_hist[c][tid] = run; run += v; }
        s_cnt[tid] = run;
    }
    __syncthreads();

    if (tid == 0) {
        s_off[0] = 0;
        for (int e = 0; e < NEXP; ++e) s_off[e + 1] = s_off[e] + s_cnt[e];
        int nt = 0;
        for (int e = 0; e < NEXP; ++e) {
            int cnt = s_cnt[e], start = s_off[e];
            for (int j = 0; j < cnt; j += 128) {
                tile_expert[nt] = e; tile_rs[nt] = start + j; tile_re[nt] = start + cnt; nt++;
            }
        }
        *num_tiles = nt;
    }
    __syncthreads();

    int run[NEXP];
    #pragma unroll
    for (int e = 0; e < NEXP; ++e) run[e] = s_hist[tid][e];
    for (int j = 0; j < 16; ++j) {
        int i = base + j, e = s_exp[i];
        int pos = s_off[e] + run[e]; run[e]++;
        src_row[pos] = i;
        dest_row[i] = pos;
    }
}

// ---------------------------------------------------------------- gather rows to bf16
__global__ __launch_bounds__(256) void gather_kernel(const float* __restrict__ x,
                                                     const int* __restrict__ src_row,
                                                     __bf16* __restrict__ xs) {
    int r = blockIdx.x;
    int t = src_row[r] >> 1;
    int c = threadIdx.x * 4;
    float4 v = *(const float4*)(x + (size_t)t * HIDDEN + c);
    bf16x4 o;
    o.x = (__bf16)v.x; o.y = (__bf16)v.y; o.z = (__bf16)v.z; o.w = (__bf16)v.w;
    *(bf16x4*)(xs + (size_t)r * HIDDEN + c) = o;
}

// ---------------------------------------------------------------- fused GEMM1 (wi_0, wi_1) + SwiGLU
// Weights pre-transposed: w0t/w1t are [E][INTER][HIDDEN] bf16 (K contiguous).
__global__ __launch_bounds__(256) void gemm1_kernel(
    const __bf16* __restrict__ xs,      // [4096][1024]
    const __bf16* __restrict__ w0t,     // [8][2048][1024]
    const __bf16* __restrict__ w1t,
    __bf16* __restrict__ inter,         // [4096][2048]
    const int* __restrict__ tile_expert, const int* __restrict__ tile_rs,
    const int* __restrict__ tile_re, const int* __restrict__ num_tiles) {
    int ty = blockIdx.y;
    if (ty >= *num_tiles) return;
    int e = tile_expert[ty];
    int rowStart = tile_rs[ty], rowEnd = tile_re[ty];
    int colBase = blockIdx.x * 128;

    __shared__ __align__(16) __bf16 sA[2][128 * 40];
    __shared__ __align__(16) __bf16 sB0[2][128 * 40];
    __shared__ __align__(16) __bf16 sB1[2][128 * 40];

    int tid = threadIdx.x;
    int lane = tid & 31;
    int wave = tid >> 5;
    int waveM = wave >> 2, waveN = wave & 3;   // 2 x 4 waves -> 64x32 per wave
    int lo = lane >> 4, ln = lane & 15;

    v8f acc0[4][2], acc1[4][2];
    #pragma unroll
    for (int m = 0; m < 4; ++m)
        #pragma unroll
        for (int n = 0; n < 2; ++n) { acc0[m][n] = (v8f)0.0f; acc1[m][n] = (v8f)0.0f; }

    const __bf16* w0e = w0t + (size_t)e * HIDDEN * INTERSZ;
    const __bf16* w1e = w1t + (size_t)e * HIDDEN * INTERSZ;

    int frow = tid >> 1;                 // 0..127
    int fseg = (tid & 1) << 4;           // 0 or 16 halves
    int arow = rowStart + frow;
    arow = arow < NROWS ? arow : NROWS - 1;
    unsigned a_go = (unsigned)(((size_t)arow * HIDDEN + fseg) * 2);
    unsigned b_go = (unsigned)(((size_t)(colBase + frow) * HIDDEN + fseg) * 2);
    unsigned l_off = (unsigned)((frow * 40 + fseg) * 2);

    auto fill = [&](int b, int kc) {
        unsigned kb = (unsigned)(kc * 2);
        async_copy32(lds_addr(sA[b])  + l_off, a_go + kb, xs);
        async_copy32(lds_addr(sB0[b]) + l_off, b_go + kb, w0e);
        async_copy32(lds_addr(sB1[b]) + l_off, b_go + kb, w1e);
    };

    fill(0, 0);
    int buf = 0;
    for (int kc = 0; kc < HIDDEN; kc += 32) {
        wait_async0();
        __syncthreads();
        if (kc + 32 < HIDDEN) fill(buf ^ 1, kc + 32);

        FragU af[4];
        #pragma unroll
        for (int mt = 0; mt < 4; ++mt) {
            const __bf16* ap = &sA[buf][(waveM * 64 + mt * 16 + ln) * 40];
            af[mt].h[0] = *(const bf16x8*)(ap + lo * 8);
            af[mt].h[1] = *(const bf16x8*)(ap + 16 + lo * 8);
        }
        #pragma unroll
        for (int nt = 0; nt < 2; ++nt) {
            int n = waveN * 32 + nt * 16 + ln;
            FragU b0f, b1f;
            b0f.h[0] = *(const bf16x8*)(&sB0[buf][n * 40 + lo * 8]);
            b0f.h[1] = *(const bf16x8*)(&sB0[buf][n * 40 + 16 + lo * 8]);
            b1f.h[0] = *(const bf16x8*)(&sB1[buf][n * 40 + lo * 8]);
            b1f.h[1] = *(const bf16x8*)(&sB1[buf][n * 40 + 16 + lo * 8]);
            #pragma unroll
            for (int mt = 0; mt < 4; ++mt) {
                acc0[mt][nt] = __builtin_amdgcn_wmma_f32_16x16x32_bf16(
                    false, af[mt].v, false, b0f.v, (short)0, acc0[mt][nt], false, false);
                acc1[mt][nt] = __builtin_amdgcn_wmma_f32_16x16x32_bf16(
                    false, af[mt].v, false, b1f.v, (short)0, acc1[mt][nt], false, false);
            }
        }
        __syncthreads();
        buf ^= 1;
    }

    #pragma unroll
    for (int mt = 0; mt < 4; ++mt)
        #pragma unroll
        for (int nt = 0; nt < 2; ++nt) {
            int col = colBase + waveN * 32 + nt * 16 + ln;
            #pragma unroll
            for (int v = 0; v < 8; ++v) {
                int row = rowStart + waveM * 64 + mt * 16 + lo * 8 + v;
                if (row < rowEnd) {
                    float h0 = acc0[mt][nt][v];
                    float h1 = acc1[mt][nt][v];
                    float s = h0 / (1.0f + __expf(-h0));   // silu
                    inter[(size_t)row * INTERSZ + col] = (__bf16)(s * h1);
                }
            }
        }
}

// ---------------------------------------------------------------- GEMM2 (wo) -> y fp32
// wot pre-transposed: [E][HIDDEN][INTER] bf16 (K contiguous).
__global__ __launch_bounds__(256) void gemm2_kernel(
    const __bf16* __restrict__ inter,   // [4096][2048]
    const __bf16* __restrict__ wot,     // [8][1024][2048]
    float* __restrict__ y,              // [4096][1024]
    const int* __restrict__ tile_expert, const int* __restrict__ tile_rs,
    const int* __restrict__ tile_re, const int* __restrict__ num_tiles) {
    int ty = blockIdx.y;
    if (ty >= *num_tiles) return;
    int e = tile_expert[ty];
    int rowStart = tile_rs[ty], rowEnd = tile_re[ty];
    int colBase = blockIdx.x * 128;

    __shared__ __align__(16) __bf16 sA[2][128 * 40];
    __shared__ __align__(16) __bf16 sB[2][128 * 40];

    int tid = threadIdx.x;
    int lane = tid & 31;
    int wave = tid >> 5;
    int waveM = wave >> 2, waveN = wave & 3;
    int lo = lane >> 4, ln = lane & 15;

    v8f acc[4][2];
    #pragma unroll
    for (int m = 0; m < 4; ++m)
        #pragma unroll
        for (int n = 0; n < 2; ++n) acc[m][n] = (v8f)0.0f;

    const __bf16* we = wot + (size_t)e * INTERSZ * HIDDEN;

    int frow = tid >> 1;
    int fseg = (tid & 1) << 4;
    int arow = rowStart + frow;
    arow = arow < NROWS ? arow : NROWS - 1;
    unsigned a_go = (unsigned)(((size_t)arow * INTERSZ + fseg) * 2);
    unsigned b_go = (unsigned)(((size_t)(colBase + frow) * INTERSZ + fseg) * 2);
    unsigned l_off = (unsigned)((frow * 40 + fseg) * 2);

    auto fill = [&](int b, int kc) {
        unsigned kb = (unsigned)(kc * 2);
        async_copy32(lds_addr(sA[b]) + l_off, a_go + kb, inter);
        async_copy32(lds_addr(sB[b]) + l_off, b_go + kb, we);
    };

    fill(0, 0);
    int buf = 0;
    for (int kc = 0; kc < INTERSZ; kc += 32) {
        wait_async0();
        __syncthreads();
        if (kc + 32 < INTERSZ) fill(buf ^ 1, kc + 32);

        FragU af[4];
        #pragma unroll
        for (int mt = 0; mt < 4; ++mt) {
            const __bf16* ap = &sA[buf][(waveM * 64 + mt * 16 + ln) * 40];
            af[mt].h[0] = *(const bf16x8*)(ap + lo * 8);
            af[mt].h[1] = *(const bf16x8*)(ap + 16 + lo * 8);
        }
        #pragma unroll
        for (int nt = 0; nt < 2; ++nt) {
            int n = waveN * 32 + nt * 16 + ln;
            FragU bf;
            bf.h[0] = *(const bf16x8*)(&sB[buf][n * 40 + lo * 8]);
            bf.h[1] = *(const bf16x8*)(&sB[buf][n * 40 + 16 + lo * 8]);
            #pragma unroll
            for (int mt = 0; mt < 4; ++mt)
                acc[mt][nt] = __builtin_amdgcn_wmma_f32_16x16x32_bf16(
                    false, af[mt].v, false, bf.v, (short)0, acc[mt][nt], false, false);
        }
        __syncthreads();
        buf ^= 1;
    }

    #pragma unroll
    for (int mt = 0; mt < 4; ++mt)
        #pragma unroll
        for (int nt = 0; nt < 2; ++nt) {
            int col = colBase + waveN * 32 + nt * 16 + ln;
            #pragma unroll
            for (int v = 0; v < 8; ++v) {
                int row = rowStart + waveM * 64 + mt * 16 + lo * 8 + v;
                if (row < rowEnd)
                    y[(size_t)row * HIDDEN + col] = acc[mt][nt][v];
            }
        }
}

// ---------------------------------------------------------------- unpermute + weighted combine
__global__ __launch_bounds__(256) void combine_kernel(const float* __restrict__ y,
                                                      const int* __restrict__ dest_row,
                                                      const float* __restrict__ topk_w,
                                                      float* __restrict__ out) {
    int t = blockIdx.x;
    int c = threadIdx.x * 4;
    int d0 = dest_row[2 * t], d1 = dest_row[2 * t + 1];
    float w0 = topk_w[2 * t], w1 = topk_w[2 * t + 1];
    float4 a = *(const float4*)(y + (size_t)d0 * HIDDEN + c);
    float4 b = *(const float4*)(y + (size_t)d1 * HIDDEN + c);
    float4 o;
    o.x = w0 * a.x + w1 * b.x;
    o.y = w0 * a.y + w1 * b.y;
    o.z = w0 * a.z + w1 * b.z;
    o.w = w0 * a.w + w1 * b.w;
    *(float4*)(out + (size_t)t * HIDDEN + c) = o;
}

// ---------------------------------------------------------------- launch
extern "C" void kernel_launch(void* const* d_in, const int* in_sizes, int n_in,
                              void* d_out, int out_size, void* d_ws, size_t ws_size,
                              hipStream_t stream) {
    const float* x      = (const float*)d_in[0];
    const float* logits = (const float*)d_in[1];
    const float* wi0    = (const float*)d_in[2];
    const float* wi1    = (const float*)d_in[3];
    const float* wo     = (const float*)d_in[4];
    float* out = (float*)d_out;

    char* p = (char*)d_ws;
    auto take = [&](size_t bytes) -> char* {
        char* r = p; p += (bytes + 255) & ~(size_t)255; return r;
    };
    const size_t WSZ = (size_t)NEXP * HIDDEN * INTERSZ;
    __bf16* w0t   = (__bf16*)take(WSZ * 2);        // [E][INTER][HIDDEN]
    __bf16* w1t   = (__bf16*)take(WSZ * 2);        // [E][INTER][HIDDEN]
    __bf16* wot   = (__bf16*)take(WSZ * 2);        // [E][HIDDEN][INTER]
    __bf16* xs    = (__bf16*)take((size_t)NROWS * HIDDEN * 2);
    __bf16* inter = (__bf16*)take((size_t)NROWS * INTERSZ * 2);
    float*  y     = (float*)take((size_t)NROWS * HIDDEN * 4);
    float*  tkw   = (float*)take(NROWS * 4);
    int* src_row  = (int*)take(NROWS * 4);
    int* dest_row = (int*)take(NROWS * 4);
    int* tile_e   = (int*)take(64 * 4);
    int* tile_rs  = (int*)take(64 * 4);
    int* tile_re  = (int*)take(64 * 4);
    int* ntiles   = (int*)take(16);

    // wi: [E][1024][2048] -> [E][2048][1024]; wo: [E][2048][1024] -> [E][1024][2048]
    transpose_cvt_kernel<<<dim3(INTERSZ / 32, HIDDEN / 32, NEXP), 256, 0, stream>>>(
        wi0, w0t, HIDDEN, INTERSZ);
    transpose_cvt_kernel<<<dim3(INTERSZ / 32, HIDDEN / 32, NEXP), 256, 0, stream>>>(
        wi1, w1t, HIDDEN, INTERSZ);
    transpose_cvt_kernel<<<dim3(HIDDEN / 32, INTERSZ / 32, NEXP), 256, 0, stream>>>(
        wo, wot, INTERSZ, HIDDEN);

    routing_kernel<<<1, 256, 0, stream>>>(logits, tkw, src_row, dest_row,
                                          tile_e, tile_rs, tile_re, ntiles);

    gather_kernel<<<NROWS, 256, 0, stream>>>(x, src_row, xs);

    gemm1_kernel<<<dim3(INTERSZ / 128, MAXTILES), 256, 0, stream>>>(
        xs, w0t, w1t, inter, tile_e, tile_rs, tile_re, ntiles);

    gemm2_kernel<<<dim3(HIDDEN / 128, MAXTILES), 256, 0, stream>>>(
        inter, wot, y, tile_e, tile_rs, tile_re, ntiles);

    combine_kernel<<<N_TOKENS, 256, 0, stream>>>(y, dest_row, tkw, out);
}
